// MultiHeadAttention_7516192768586
// MI455X (gfx1250) — matrix operand order repaired and checked by reading, not compiled, not run on hardware
//
#include <hip/hip_runtime.h>
#include <hip/hip_bf16.h>

// ---------- types ----------
typedef __attribute__((ext_vector_type(16))) _Float16 v16h;
typedef __attribute__((ext_vector_type(8)))  float    v8f;
typedef __attribute__((ext_vector_type(4)))  float    v4f;

#define TSEQ   2048
#define DMODEL 1024
#define NBATCH 2
#define DEPTH  64
#define PADF   (-4294967295.0f)
#define NEGBIG (-3.402823466e38f)

union H16u { v16h h; v4f f4[2]; _Float16 e[16]; };

static __device__ __forceinline__ v16h ld2x16(const _Float16* lo, const _Float16* hi) {
  H16u u;
  u.f4[0] = *(const v4f*)lo;   // 16B aligned chunk: elements 0..7
  u.f4[1] = *(const v4f*)hi;   // elements 8..15
  return u.h;
}
static __device__ __forceinline__ v16h ld32(const _Float16* p) {
  return ld2x16(p, p + 8);     // contiguous 32B (>=16B aligned)
}

// ---------- kernel 1: f32 -> f16 convert ----------
__global__ void cvt_f32_f16(const float* __restrict__ src, _Float16* __restrict__ dst, int n) {
  int i = blockIdx.x * blockDim.x + threadIdx.x;
  if (i < n) dst[i] = (_Float16)src[i];
}

// ---------- kernel 2: projection GEMM  Y = X * W^T + b, head-major f16 out ----------
// grid(16, 64, 3) block(128): 4 waves, wave w -> rows [by*64 + w*16, +16), cols [bx*64, +64)
__global__ void proj_gemm(const _Float16* __restrict__ Xq, const _Float16* __restrict__ Xk,
                          const _Float16* __restrict__ Xv,
                          const _Float16* __restrict__ Wq, const _Float16* __restrict__ Wk,
                          const _Float16* __restrict__ Wv,
                          const float* __restrict__ Bq, const float* __restrict__ Bk,
                          const float* __restrict__ Bv,
                          _Float16* __restrict__ Yq, _Float16* __restrict__ Yk,
                          _Float16* __restrict__ Yv) {
  const int z = blockIdx.z;
  const _Float16* X = (z == 0) ? Xq : (z == 1) ? Xk : Xv;
  const _Float16* W = (z == 0) ? Wq : (z == 1) ? Wk : Wv;
  const float*    B = (z == 0) ? Bq : (z == 1) ? Bk : Bv;
  _Float16*       Y = (z == 0) ? Yq : (z == 1) ? Yk : Yv;

  const int lane = threadIdx.x & 31;
  const int wave = threadIdx.x >> 5;
  const int g    = lane >> 4;      // half-wave group
  const int ln   = lane & 15;

  const int n0 = blockIdx.x * 64;
  const int m0 = blockIdx.y * 64 + wave * 16;

  v8f acc[4] = {};
  const _Float16* xrow = X + (size_t)(m0 + ln) * DMODEL;

  for (int k0 = 0; k0 < DMODEL; k0 += 32) {
    // A 16x32: lane(g,ln) row=ln, K = 16g..16g+15
    v16h a = ld2x16(xrow + k0 + 16 * g, xrow + k0 + 16 * g + 8);
#pragma unroll
    for (int nt = 0; nt < 4; ++nt) {
      // B 32x16: lane(g,ln) col = ln, K = k0+16g .. +15  (contiguous W row)
      const _Float16* wrow = W + (size_t)(n0 + nt * 16 + ln) * DMODEL + k0 + 16 * g;
      v16h b = ld32(wrow);
      acc[nt] = __builtin_amdgcn_wmma_f32_16x16x32_f16(false, a, false, b,
                                                       (short)0, acc[nt], false, false);
    }
  }

#pragma unroll
  for (int nt = 0; nt < 4; ++nt) {
    const int col  = n0 + nt * 16 + ln;
    const float bv = B[col];
    const int head = col >> 6;
    const int dl   = col & 63;
#pragma unroll
    for (int r = 0; r < 8; ++r) {
      const int m  = m0 + r + 8 * g;      // global row in (B*T)
      const int bb = m >> 11;             // T = 2048
      const int t  = m & (TSEQ - 1);
      Y[((size_t)(head * NBATCH + bb) * TSEQ + t) * DEPTH + dl] = (_Float16)(acc[nt][r] + bv);
    }
  }
}

// ---------- kernel 3: flash attention with async K staging ----------
// grid(32, 32) block(128). blockIdx.x = head*B+b, wave handles 16 query rows.
__global__ void attn_flash(const _Float16* __restrict__ qh, const _Float16* __restrict__ kh,
                           const _Float16* __restrict__ vh, const int* __restrict__ mask,
                           const int* __restrict__ causp, float* __restrict__ ctxbuf) {
  // K block double buffer, row-major [s][d], rows padded to 72 halves (144B) for bank spread
  __shared__ __attribute__((aligned(16))) _Float16 Krow[2][32][72];
  __shared__ __attribute__((aligned(16))) _Float16 Vt[64][40];       // [d][s], padded
  __shared__ __attribute__((aligned(16))) _Float16 Psh[4][16][32];   // per-wave P tile

  const int lane = threadIdx.x & 31;
  const int wave = threadIdx.x >> 5;
  const int g    = lane >> 4;
  const int ln   = lane & 15;

  const int nhb  = blockIdx.x;          // head*B + b
  const int b    = nhb & (NBATCH - 1);
  const int head = nhb >> 1;
  const int t0   = blockIdx.y * 64 + wave * 16;
  const int caus = causp[0];

  // per-thread staging coordinates: 32 rows x 4 chunks of 32B
  const int sl = threadIdx.x >> 2;          // 0..31 (key row within block)
  const int dc = (threadIdx.x & 3) * 16;    // 0,16,32,48 (depth chunk)

  // prologue: async-stage K block 0 into buffer 0 (2 x b128 = 32B per thread)
  {
    const _Float16* gsrc = kh + ((size_t)nhb * TSEQ + sl) * DEPTH + dc;
    unsigned       ldsa = (unsigned)(uintptr_t)&Krow[0][sl][dc];   // low 32b = LDS offset
    unsigned long long ga = (unsigned long long)(uintptr_t)gsrc;
    asm volatile("global_load_async_to_lds_b128 %0, %1, off"
                 :: "v"(ldsa), "v"(ga) : "memory");
    asm volatile("global_load_async_to_lds_b128 %0, %1, off offset:16"
                 :: "v"(ldsa), "v"(ga) : "memory");
  }

  // Q A-operands for K=0..31 and K=32..63, kept in registers
  const _Float16* qrow = qh + ((size_t)nhb * TSEQ + (t0 + ln)) * DEPTH;
  const v16h qa0 = ld2x16(qrow + 16 * g,      qrow + 8 + 16 * g);
  const v16h qa1 = ld2x16(qrow + 32 + 16 * g, qrow + 40 + 16 * g);

  float mrun[8], lrun[8];
  v8f acc[4] = {};
#pragma unroll
  for (int r = 0; r < 8; ++r) { mrun[r] = NEGBIG; lrun[r] = 0.0f; }

  for (int s0 = 0; s0 < TSEQ; s0 += 32) {
    const int bi = (s0 >> 5) & 1;
    __syncthreads();  // protect Vt / Krow[bi^1] (WAR vs previous iteration's readers)

    {  // stage V block 32(s) x 64(d), transposed into Vt[d][s]; prefetch next block
      H16u u;
      u.h = ld32(vh + ((size_t)nhb * TSEQ + s0 + sl) * DEPTH + dc);
#pragma unroll
      for (int j = 0; j < 16; ++j) Vt[dc + j][sl] = u.e[j];
      if (s0 + 32 < TSEQ)
        __builtin_prefetch(vh + ((size_t)nhb * TSEQ + s0 + 32 + sl) * DEPTH + dc, 0, 0);
    }

    // issue async K stage for next block into alternate buffer, then wait for current.
    // async loads complete in order, so asynccnt<=2 means the current block landed.
    if (s0 + 32 < TSEQ) {
      const _Float16* gsrc = kh + ((size_t)nhb * TSEQ + s0 + 32 + sl) * DEPTH + dc;
      unsigned       ldsa = (unsigned)(uintptr_t)&Krow[bi ^ 1][sl][dc];
      unsigned long long ga = (unsigned long long)(uintptr_t)gsrc;
      asm volatile("global_load_async_to_lds_b128 %0, %1, off"
                   :: "v"(ldsa), "v"(ga) : "memory");
      asm volatile("global_load_async_to_lds_b128 %0, %1, off offset:16"
                   :: "v"(ldsa), "v"(ga) : "memory");
      asm volatile("s_wait_asynccnt 0x2" ::: "memory");
    } else {
      asm volatile("s_wait_asynccnt 0x0" ::: "memory");
    }
    __syncthreads();

    // scores: two 16x16 tiles (keys s0..s0+15, s0+16..s0+31), K = depth = 64
    v8f sc[2];
#pragma unroll
    for (int st = 0; st < 2; ++st) {
      const _Float16* krow = &Krow[bi][st * 16 + ln][16 * g];   // col = key, K = d
      v16h kb0 = ld2x16(krow,      krow + 8);       // d = 16g .. 16g+15
      v16h kb1 = ld2x16(krow + 32, krow + 40);      // d = 32+16g .. +15
      v8f c = {};
      c = __builtin_amdgcn_wmma_f32_16x16x32_f16(false, qa0, false, kb0, (short)0, c, false, false);
      c = __builtin_amdgcn_wmma_f32_16x16x32_f16(false, qa1, false, kb1, (short)0, c, false, false);
      sc[st] = c;
    }

    // scale + key mask (+ causal), per-lane column addend
    const float pad0 = mask[b * TSEQ + s0 + ln]      ? PADF : 0.0f;
    const float pad1 = mask[b * TSEQ + s0 + 16 + ln] ? PADF : 0.0f;
#pragma unroll
    for (int r = 0; r < 8; ++r) {
      const int t = t0 + r + 8 * g;
      float v0 = sc[0][r] * 0.125f + pad0;   // 1/sqrt(64)
      float v1 = sc[1][r] * 0.125f + pad1;
      if (caus) {
        v0 = ((s0 + ln)      > t) ? PADF : v0;
        v1 = ((s0 + 16 + ln) > t) ? PADF : v1;
      }
      sc[0][r] = v0; sc[1][r] = v1;
    }

    // online softmax update; write P (f16) to wave-private LDS for D->A relayout
    float alpha[8];
#pragma unroll
    for (int r = 0; r < 8; ++r) {
      float bm = fmaxf(sc[0][r], sc[1][r]);
      bm = fmaxf(bm, __shfl_xor(bm, 8, 32));
      bm = fmaxf(bm, __shfl_xor(bm, 4, 32));
      bm = fmaxf(bm, __shfl_xor(bm, 2, 32));
      bm = fmaxf(bm, __shfl_xor(bm, 1, 32));
      const float mn = fmaxf(mrun[r], bm);
      const float a  = __expf(mrun[r] - mn);
      const float p0 = __expf(sc[0][r] - mn);
      const float p1 = __expf(sc[1][r] - mn);
      float rs = p0 + p1;
      rs += __shfl_xor(rs, 8, 32);
      rs += __shfl_xor(rs, 4, 32);
      rs += __shfl_xor(rs, 2, 32);
      rs += __shfl_xor(rs, 1, 32);
      lrun[r]  = lrun[r] * a + rs;
      mrun[r]  = mn;
      alpha[r] = a;
      Psh[wave][r + 8 * g][ln]      = (_Float16)p0;   // element (m, s_local)
      Psh[wave][r + 8 * g][16 + ln] = (_Float16)p1;
    }
#pragma unroll
    for (int dt = 0; dt < 4; ++dt)
#pragma unroll
      for (int r = 0; r < 8; ++r) acc[dt][r] *= alpha[r];

    // P(16x32) x V(32x64): A from LDS relayout, B from transposed Vt (contiguous)
    const _Float16* prow = &Psh[wave][ln][0];
    const v16h pa = ld2x16(prow + 16 * g, prow + 8 + 16 * g);
#pragma unroll
    for (int dt = 0; dt < 4; ++dt) {
      const _Float16* vrow = &Vt[dt * 16 + ln][16 * g];  // col = dt*16+ln, K = 16g..+15
      v16h vb = ld2x16(vrow, vrow + 8);
      acc[dt] = __builtin_amdgcn_wmma_f32_16x16x32_f16(false, pa, false, vb,
                                                       (short)0, acc[dt], false, false);
    }
  }

  // normalize and scatter to merged-head f32 buffer (b, t, head*64 + d)
#pragma unroll
  for (int r = 0; r < 8; ++r) {
    const float inv = 1.0f / lrun[r];
    const int t = t0 + r + 8 * g;
    float* orow = ctxbuf + ((size_t)b * TSEQ + t) * DMODEL + head * DEPTH;
#pragma unroll
    for (int dt = 0; dt < 4; ++dt) orow[dt * 16 + ln] = acc[dt][r] * inv;
  }
}

// ---------- kernel 4: residual + LayerNorm ----------
__global__ void resid_ln(const float* __restrict__ ctxbuf, const float* __restrict__ qin,
                         const float* __restrict__ gamma, const float* __restrict__ beta,
                         float* __restrict__ out) {
  const int row = blockIdx.x;
  const int tid = threadIdx.x;          // 256 threads, 4 elems each
  __shared__ float s1[256], s2[256];
  float x[4];
  float sum = 0.0f, sq = 0.0f;
#pragma unroll
  for (int i = 0; i < 4; ++i) {
    const int c = tid + i * 256;
    const float v = ctxbuf[(size_t)row * DMODEL + c] + qin[(size_t)row * DMODEL + c];
    x[i] = v; sum += v; sq += v * v;
  }
  s1[tid] = sum; s2[tid] = sq;
  __syncthreads();
  for (int o = 128; o > 0; o >>= 1) {
    if (tid < o) { s1[tid] += s1[tid + o]; s2[tid] += s2[tid + o]; }
    __syncthreads();
  }
  const float mean = s1[0] * (1.0f / DMODEL);
  const float var  = s2[0] * (1.0f / DMODEL) - mean * mean;
  const float rstd = rsqrtf(var + 1e-5f);
#pragma unroll
  for (int i = 0; i < 4; ++i) {
    const int c = tid + i * 256;
    out[(size_t)row * DMODEL + c] = (x[i] - mean) * rstd * gamma[c] + beta[c];
  }
}

// ---------- launcher ----------
extern "C" void kernel_launch(void* const* d_in, const int* in_sizes, int n_in,
                              void* d_out, int out_size, void* d_ws, size_t ws_size,
                              hipStream_t stream) {
  const float* q    = (const float*)d_in[0];
  const float* k    = (const float*)d_in[1];
  const float* v    = (const float*)d_in[2];
  const int*   mask = (const int*)d_in[3];
  const int*   caus = (const int*)d_in[4];
  const float* wq   = (const float*)d_in[6];
  const float* bq   = (const float*)d_in[7];
  const float* wk   = (const float*)d_in[8];
  const float* bk   = (const float*)d_in[9];
  const float* wv   = (const float*)d_in[10];
  const float* bv   = (const float*)d_in[11];
  const float* gamma = (const float*)d_in[12];
  const float* beta  = (const float*)d_in[13];
  float* out = (float*)d_out;

  const size_t MB = 1u << 20;
  char* ws = (char*)d_ws;
  _Float16* Xq16 = (_Float16*)(ws + 0 * MB);
  _Float16* Xk16 = (_Float16*)(ws + 8 * MB);
  _Float16* Xv16 = (_Float16*)(ws + 16 * MB);
  _Float16* Wq16 = (_Float16*)(ws + 24 * MB);
  _Float16* Wk16 = (_Float16*)(ws + 26 * MB);
  _Float16* Wv16 = (_Float16*)(ws + 28 * MB);
  _Float16* qh   = (_Float16*)(ws + 30 * MB);
  _Float16* kh   = (_Float16*)(ws + 38 * MB);
  _Float16* vh   = (_Float16*)(ws + 46 * MB);
  float*    ctx  = (float*)   (ws + 54 * MB);   // ends at 70 MB

  const int NX = NBATCH * TSEQ * DMODEL;   // 4,194,304
  const int NW = DMODEL * DMODEL;          // 1,048,576
  cvt_f32_f16<<<(NX + 255) / 256, 256, 0, stream>>>(q, Xq16, NX);
  cvt_f32_f16<<<(NX + 255) / 256, 256, 0, stream>>>(k, Xk16, NX);
  cvt_f32_f16<<<(NX + 255) / 256, 256, 0, stream>>>(v, Xv16, NX);
  cvt_f32_f16<<<(NW + 255) / 256, 256, 0, stream>>>(wq, Wq16, NW);
  cvt_f32_f16<<<(NW + 255) / 256, 256, 0, stream>>>(wk, Wk16, NW);
  cvt_f32_f16<<<(NW + 255) / 256, 256, 0, stream>>>(wv, Wv16, NW);

  proj_gemm<<<dim3(DMODEL / 64, (NBATCH * TSEQ) / 64, 3), 128, 0, stream>>>(
      Xq16, Xk16, Xv16, Wq16, Wk16, Wv16, bq, bk, bv, qh, kh, vh);

  attn_flash<<<dim3(16 * NBATCH, TSEQ / 64), 128, 0, stream>>>(qh, kh, vh, mask, caus, ctx);

  resid_ln<<<NBATCH * TSEQ, 256, 0, stream>>>(ctx, q, gamma, beta, out);
}
